// InterpretableFusion_59682865545265
// MI455X (gfx1250) — matrix operand order repaired and checked by reading, not compiled, not run on hardware
//
#include <hip/hip_runtime.h>
#include <cstdint>
#include <cstddef>

// MI455X / gfx1250, wave32. WMMA bf16 16x16x32 with bf16x3 split-precision.
typedef __attribute__((ext_vector_type(16))) __bf16 v16bf;
typedef __attribute__((ext_vector_type(8)))  float  v8f;
typedef __attribute__((ext_vector_type(4)))  __bf16 v4bf;

#define D_DIM 512
#define M_MOD 4
#define TN    128          // rows per workgroup tile
#define A_ELEMS (8*16*32*16)   // [blockRow][kchunk][lane][16] bf16 = 65536 elems (128KB)

static __device__ __forceinline__ void split_bf16(float x, __bf16& h, __bf16& l) {
  h = (__bf16)x;
  l = (__bf16)(x - (float)h);
}

// ---------------------------------------------------------------------------
// Kernel 1: CallT[col][k] = sum_e Wq[e][k] * Wk[m][e][col%512],  col = m*512+d'
// stored column-major in k (so WMMA B-fragments are contiguous 32B per lane),
// split into bf16 hi/lo. 64x64 output tile per WG, LDS-staged f32 GEMM.
// ---------------------------------------------------------------------------
__global__ void __launch_bounds__(256)
k_precompute_callT(const float* __restrict__ Wq, const float* __restrict__ Wk,
                   __bf16* __restrict__ CTh, __bf16* __restrict__ CTl) {
  __shared__ float As[16][64];
  __shared__ float Bs[16][64];
  const int m  = blockIdx.z;
  const int kT = blockIdx.x * 64;
  const int cT = blockIdx.y * 64;
  const int tl = threadIdx.x;
  const int ty = tl >> 4, tx = tl & 15;
  const int er = tl >> 6, ec = tl & 63;
  const float* wkm = Wk + (size_t)m * D_DIM * D_DIM;

  float acc[4][4];
#pragma unroll
  for (int i = 0; i < 4; ++i)
#pragma unroll
    for (int j = 0; j < 4; ++j) acc[i][j] = 0.f;

  for (int e0 = 0; e0 < D_DIM; e0 += 16) {
#pragma unroll
    for (int p = 0; p < 4; ++p) {
      As[er + 4*p][ec] = Wq [(size_t)(e0 + er + 4*p) * D_DIM + kT + ec];
      Bs[er + 4*p][ec] = wkm[(size_t)(e0 + er + 4*p) * D_DIM + cT + ec];
    }
    __syncthreads();
#pragma unroll
    for (int e = 0; e < 16; ++e) {
      float a[4], b[4];
#pragma unroll
      for (int i = 0; i < 4; ++i) a[i] = As[e][ty*4 + i];
#pragma unroll
      for (int j = 0; j < 4; ++j) b[j] = Bs[e][tx*4 + j];
#pragma unroll
      for (int i = 0; i < 4; ++i)
#pragma unroll
        for (int j = 0; j < 4; ++j) acc[i][j] += a[i] * b[j];
    }
    __syncthreads();
  }
#pragma unroll
  for (int i = 0; i < 4; ++i)
#pragma unroll
    for (int j = 0; j < 4; ++j) {
      size_t col = (size_t)m * D_DIM + cT + tx*4 + j;
      size_t k   = (size_t)kT + ty*4 + i;
      __bf16 h, l; split_bf16(acc[i][j], h, l);
      CTh[col * D_DIM + k] = h;
      CTl[col * D_DIM + k] = l;
    }
}

// ---------------------------------------------------------------------------
// Kernel 2: aux vectors. tb[m][d'] = bq^T Wk[m];  u[m][d] = Wq^T bk[m]
// (stored as 16-col padded bf16 hi/lo B-matrix "UT");  c[m] = bq.bk[m]
// ---------------------------------------------------------------------------
__global__ void __launch_bounds__(512)
k_precompute_aux(const float* __restrict__ Wq, const float* __restrict__ bq,
                 const float* __restrict__ Wk, const float* __restrict__ bk,
                 __bf16* __restrict__ UTh, __bf16* __restrict__ UTl,
                 float* __restrict__ tb, float* __restrict__ cvec) {
  const int b = blockIdx.x, tid = threadIdx.x;
  if (b < 4) {
    const int m = b;
    const float* wkm = Wk + (size_t)m * D_DIM * D_DIM;
    const float* bkm = bk + (size_t)m * D_DIM;
    float at = 0.f, au = 0.f;
    for (int e = 0; e < D_DIM; ++e) {
      at += bq[e] * wkm[(size_t)e * D_DIM + tid];
      au += Wq[(size_t)e * D_DIM + tid] * bkm[e];
    }
    tb[m * D_DIM + tid] = at;
    __bf16 h, l; split_bf16(au, h, l);
    UTh[(size_t)m * D_DIM + tid] = h;
    UTl[(size_t)m * D_DIM + tid] = l;
  } else {
    for (int i = tid; i < 12 * D_DIM; i += 512) {     // zero pad cols 4..15
      UTh[4 * D_DIM + i] = (__bf16)0.f;
      UTl[4 * D_DIM + i] = (__bf16)0.f;
    }
    if (tid < 4) {
      const float* bkm = bk + (size_t)tid * D_DIM;
      float a = 0.f;
      for (int e = 0; e < D_DIM; ++e) a += bq[e] * bkm[e];
      cvec[tid] = a;
    }
  }
}

// ---------------------------------------------------------------------------
// Kernel 3: fused main. Per WG: 128 rows.
//  Phase 0: stage gnn tile into LDS, pre-swizzled into WMMA A-fragment order
//           (bf16 hi + lo).  Phase 1: T = gnn @ Call via v_wmma_f32_16x16x32_bf16
//           (3-term split), epilogue dots each 16-col T chunk with latents ->
//           scores (LDS atomics).  Extra chunk: gnn.u[m].  Phase 2: softmax +
//           weighted latent sum.
// ---------------------------------------------------------------------------
__global__ void __launch_bounds__(256, 1)
k_fusion_main(const float* __restrict__ gnn, const float* __restrict__ lat,
              const __bf16* __restrict__ CTh, const __bf16* __restrict__ CTl,
              const __bf16* __restrict__ UTh, const __bf16* __restrict__ UTl,
              const float* __restrict__ tb, const float* __restrict__ cvec,
              float* __restrict__ out_f, float* __restrict__ out_w, int N) {
  extern __shared__ char smem[];
  __bf16* Ahi = (__bf16*)smem;                 // [8][16][32][16]
  __bf16* Alo = Ahi + A_ELEMS;
  float*  sco = (float*)(smem + 2 * A_ELEMS * 2);   // 128*4 scores
  float*  wts = sco + TN * 4;                       // 128*4 softmax weights

  const int tid = threadIdx.x;
  const int r0  = blockIdx.x * TN;

  sco[tid]       = cvec[tid & 3];
  sco[tid + 256] = cvec[(tid + 256) & 3];

  // ---- Phase 0: load gnn tile (coalesced float4) -> swizzled bf16 hi/lo LDS
#pragma unroll 4
  for (int i = 0; i < 64; ++i) {
    int idx = tid + 256 * i;            // 128 rows x 128 float4
    int row = idx >> 7, d4 = idx & 127;
    float4 g = make_float4(0.f, 0.f, 0.f, 0.f);
    if (r0 + row < N) g = *(const float4*)(gnn + (size_t)(r0 + row) * D_DIM + d4 * 4);
    int d   = d4 * 4;
    int br  = row >> 4, lr = row & 15;
    int kc  = d >> 5,  kin = d & 31;
    int ln_ = lr + ((kin >> 4) << 4);             // lane within fragment
    int pos = kin & 15;                           // 0,4,8,12 -> 8B aligned
    int off = (((br * 16 + kc) * 32) + ln_) * 16 + pos;
    __bf16 h0,h1,h2,h3,l0,l1,l2,l3;
    split_bf16(g.x, h0, l0); split_bf16(g.y, h1, l1);
    split_bf16(g.z, h2, l2); split_bf16(g.w, h3, l3);
    v4bf hv = {h0, h1, h2, h3}, lv = {l0, l1, l2, l3};
    *(v4bf*)(Ahi + off) = hv;
    *(v4bf*)(Alo + off) = lv;
  }
  __syncthreads();

  // ---- Phase 1: WMMA GEMM + fused score epilogue
  const int wave = tid >> 5;
  const int lane = tid & 31;
  const int lh   = lane >> 4;     // lane half
  const int ln   = lane & 15;     // column within 16-wide chunk
  const v8f vzero = {0.f,0.f,0.f,0.f,0.f,0.f,0.f,0.f};

  for (int m = 0; m < M_MOD; ++m) {
    float psum[8][8];
#pragma unroll
    for (int a = 0; a < 8; ++a)
#pragma unroll
      for (int b = 0; b < 8; ++b) psum[a][b] = 0.f;

    for (int t = 0; t < 4; ++t) {
      const int c   = 32 * m + 8 * t + wave;   // column chunk 0..127
      const int dpb = (c & 31) * 16;           // d' base within modality m
      const size_t bbase = ((size_t)(m * D_DIM + dpb + ln)) * D_DIM + (lh << 4);
      v8f acc[8];
#pragma unroll
      for (int br = 0; br < 8; ++br) acc[br] = vzero;

#pragma unroll 2
      for (int kc = 0; kc < 16; ++kc) {
        v16bf bhi = *(const v16bf*)(CTh + bbase + kc * 32);
        v16bf blo = *(const v16bf*)(CTl + bbase + kc * 32);
#pragma unroll
        for (int br = 0; br < 8; ++br) {
          const __bf16* ap = Ahi + (((br * 16 + kc) * 32) + lane) * 16;
          v16bf ahi = *(const v16bf*)ap;
          v16bf alo = *(const v16bf*)(ap + A_ELEMS);
          acc[br] = __builtin_amdgcn_wmma_f32_16x16x32_bf16(false, ahi, false, bhi, (short)0, acc[br], false, false);
          acc[br] = __builtin_amdgcn_wmma_f32_16x16x32_bf16(false, ahi, false, blo, (short)0, acc[br], false, false);
          acc[br] = __builtin_amdgcn_wmma_f32_16x16x32_bf16(false, alo, false, bhi, (short)0, acc[br], false, false);
        }
      }
      // epilogue: psum[row] += lat[n,m,d'] * (T + tb[m,d'])
      const float tbv = tb[m * D_DIM + dpb + ln];
#pragma unroll
      for (int br = 0; br < 8; ++br)
#pragma unroll
        for (int v = 0; v < 8; ++v) {
          int row = br * 16 + v + (lh << 3);
          float tval = acc[br][v] + tbv;
          float lv = 0.f;
          if (r0 + row < N)
            lv = lat[((size_t)(r0 + row) * M_MOD + m) * D_DIM + dpb + ln];
          psum[br][v] += lv * tval;
        }
    }
    // reduce across the 16 lanes of each half, accumulate into LDS scores
#pragma unroll
    for (int br = 0; br < 8; ++br)
#pragma unroll
      for (int v = 0; v < 8; ++v) {
        float p = psum[br][v];
        p += __shfl_xor(p, 1);
        p += __shfl_xor(p, 2);
        p += __shfl_xor(p, 4);
        p += __shfl_xor(p, 8);
        if (ln == 0) {
          int row = br * 16 + v + (lh << 3);
          atomicAdd(&sco[row * 4 + m], p);
        }
      }
  }

  // ---- extra chunk (wave 0): scores[n,m] += gnn[n].u[m]
  if (wave == 0) {
    v8f acc[8];
#pragma unroll
    for (int br = 0; br < 8; ++br) acc[br] = vzero;
    const size_t ub = (size_t)ln * D_DIM + (lh << 4);
#pragma unroll 2
    for (int kc = 0; kc < 16; ++kc) {
      v16bf bhi = *(const v16bf*)(UTh + ub + kc * 32);
      v16bf blo = *(const v16bf*)(UTl + ub + kc * 32);
#pragma unroll
      for (int br = 0; br < 8; ++br) {
        const __bf16* ap = Ahi + (((br * 16 + kc) * 32) + lane) * 16;
        v16bf ahi = *(const v16bf*)ap;
        v16bf alo = *(const v16bf*)(ap + A_ELEMS);
        acc[br] = __builtin_amdgcn_wmma_f32_16x16x32_bf16(false, ahi, false, bhi, (short)0, acc[br], false, false);
        acc[br] = __builtin_amdgcn_wmma_f32_16x16x32_bf16(false, ahi, false, blo, (short)0, acc[br], false, false);
        acc[br] = __builtin_amdgcn_wmma_f32_16x16x32_bf16(false, alo, false, bhi, (short)0, acc[br], false, false);
      }
    }
    if (ln < M_MOD) {
#pragma unroll
      for (int br = 0; br < 8; ++br)
#pragma unroll
        for (int v = 0; v < 8; ++v) {
          int row = br * 16 + v + (lh << 3);
          atomicAdd(&sco[row * 4 + ln], acc[br][v]);
        }
    }
  }
  __syncthreads();

  // ---- Phase 2a: softmax over m (scale 1/sqrt(D)), write weights output
  if (tid < TN) {
    const float sc = 0.04419417382415922f;   // 1/sqrt(512)
    float s0 = sco[tid * 4 + 0] * sc, s1 = sco[tid * 4 + 1] * sc;
    float s2 = sco[tid * 4 + 2] * sc, s3 = sco[tid * 4 + 3] * sc;
    float mx = fmaxf(fmaxf(s0, s1), fmaxf(s2, s3));
    float e0 = __expf(s0 - mx), e1 = __expf(s1 - mx);
    float e2 = __expf(s2 - mx), e3 = __expf(s3 - mx);
    float inv = 1.f / (e0 + e1 + e2 + e3);
    float w0 = e0 * inv, w1 = e1 * inv, w2 = e2 * inv, w3 = e3 * inv;
    wts[tid * 4 + 0] = w0; wts[tid * 4 + 1] = w1;
    wts[tid * 4 + 2] = w2; wts[tid * 4 + 3] = w3;
    if (r0 + tid < N)
      *(float4*)(out_w + (size_t)(r0 + tid) * 4) = make_float4(w0, w1, w2, w3);
  }
  __syncthreads();

  // ---- Phase 2b: fused[n,d] = sum_m w[n,m] * lat[n,m,d]  (latents hit L2)
#pragma unroll 2
  for (int i = 0; i < 64; ++i) {
    int idx = tid + 256 * i;
    int row = idx >> 7, d4 = idx & 127;
    if (r0 + row < N) {
      float w0 = wts[row * 4 + 0], w1 = wts[row * 4 + 1];
      float w2 = wts[row * 4 + 2], w3 = wts[row * 4 + 3];
      const float* lp = lat + (size_t)(r0 + row) * M_MOD * D_DIM + d4 * 4;
      float4 L0 = *(const float4*)(lp);
      float4 L1 = *(const float4*)(lp + D_DIM);
      float4 L2 = *(const float4*)(lp + 2 * D_DIM);
      float4 L3 = *(const float4*)(lp + 3 * D_DIM);
      float4 r;
      r.x = w0 * L0.x + w1 * L1.x + w2 * L2.x + w3 * L3.x;
      r.y = w0 * L0.y + w1 * L1.y + w2 * L2.y + w3 * L3.y;
      r.z = w0 * L0.z + w1 * L1.z + w2 * L2.z + w3 * L3.z;
      r.w = w0 * L0.w + w1 * L1.w + w2 * L2.w + w3 * L3.w;
      *(float4*)(out_f + (size_t)(r0 + row) * D_DIM + d4 * 4) = r;
    }
  }
}

// ---------------------------------------------------------------------------
extern "C" void kernel_launch(void* const* d_in, const int* in_sizes, int n_in,
                              void* d_out, int out_size, void* d_ws, size_t ws_size,
                              hipStream_t stream) {
  const float* gnn = (const float*)d_in[0];
  const float* lat = (const float*)d_in[1];
  const float* Wq  = (const float*)d_in[2];
  const float* bq  = (const float*)d_in[3];
  const float* Wk  = (const float*)d_in[4];
  const float* bk  = (const float*)d_in[5];
  const int N = in_sizes[0] / D_DIM;

  float* out_f = (float*)d_out;
  float* out_w = out_f + (size_t)N * D_DIM;

  char* ws = (char*)d_ws;
  __bf16* CTh = (__bf16*)ws;                       // 2048*512 bf16
  __bf16* CTl = CTh + (size_t)2048 * 512;
  __bf16* UTh = CTl + (size_t)2048 * 512;          // 16*512 bf16
  __bf16* UTl = UTh + 16 * 512;
  float*  tbp = (float*)(UTl + 16 * 512);          // 4*512 f32
  float*  cvp = tbp + 4 * 512;                     // 4 f32

  hipLaunchKernelGGL(k_precompute_callT, dim3(8, 8, 4), dim3(256), 0, stream,
                     Wq, Wk, CTh, CTl);
  hipLaunchKernelGGL(k_precompute_aux, dim3(5), dim3(512), 0, stream,
                     Wq, bq, Wk, bk, UTh, UTl, tbp, cvp);

  const int nblk = (N + TN - 1) / TN;
  const size_t shbytes = (size_t)2 * A_ELEMS * 2 + 2 * TN * 4 * sizeof(float); // 266240
  hipFuncSetAttribute((const void*)k_fusion_main,
                      hipFuncAttributeMaxDynamicSharedMemorySize, (int)shbytes);
  hipLaunchKernelGGL(k_fusion_main, dim3(nblk), dim3(256), shbytes, stream,
                     gnn, lat, CTh, CTl, UTh, UTl, tbp, cvp, out_f, out_w, N);
}
